// MultiHeadAttention_23450521436544
// MI455X (gfx1250) — matrix operand order repaired and checked by reading, not compiled, not run on hardware
//
#include <hip/hip_runtime.h>
#include <hip/hip_bf16.h>
#include <stdint.h>

typedef __attribute__((ext_vector_type(16))) _Float16 v16h;
typedef __attribute__((ext_vector_type(8)))  _Float16 v8h;
typedef __attribute__((ext_vector_type(8)))  float    v8f;

#define D_MODEL   1024
#define NUM_HEADS 16
#define D_HEAD    64
#define SEQ       2048
#define BATCH     2
#define MROWS     (BATCH * SEQ)   // 4096

// ---- gfx1250 async global->LDS path (ASYNCcnt-tracked DMA, no VGPR round trip)
#if defined(__has_builtin)
#  if __has_builtin(__builtin_amdgcn_global_load_async_to_lds_b128)
#    define MHA_ASYNC 1
#  endif
#endif
#ifndef MHA_ASYNC
#  define MHA_ASYNC 0
#endif

#if MHA_ASYNC
// The builtin's parameters are pointers to a 16-byte int vector in AS1 / AS3
// (clang diagnostic: "__attribute__((__vector_size__(4*sizeof(int)))) int
// __device__ *"). On AMDGPU the low 32 bits of a flat shared address are the
// LDS byte offset, and flat global addresses equal the global address, so
// integer round-trips produce valid segment pointers.
typedef __attribute__((vector_size(16))) int v4i;
typedef __attribute__((address_space(1))) v4i* gv4i_p;
typedef __attribute__((address_space(3))) v4i* lv4i_p;

__device__ __forceinline__ lv4i_p to_lds(void* p) {
  return (lv4i_p)(uint32_t)(uintptr_t)p;
}
__device__ __forceinline__ gv4i_p to_glob(const void* p) {
  return (gv4i_p)(uintptr_t)p;
}
#endif

// Stage one contiguous 8KB f16 tile (512 x 16B) into LDS.
__device__ __forceinline__ void stage_tile(_Float16* dst, const _Float16* src,
                                           int tid) {
#if MHA_ASYNC
#pragma unroll
  for (int i = 0; i < 4; ++i) {
    int idx = tid + i * 128;
    __builtin_amdgcn_global_load_async_to_lds_b128(
        to_glob(src + idx * 8), to_lds(dst + idx * 8), 0, 0);
  }
#else
  const int4* s = (const int4*)src;
  int4* d = (int4*)dst;
#pragma unroll
  for (int i = 0; i < 4; ++i) d[tid + i * 128] = s[tid + i * 128];
#endif
}

// Wait until at most N of this wave's async tile-copies remain outstanding.
__device__ __forceinline__ void wait_async_le4() {
#if MHA_ASYNC
  asm volatile("s_wait_asynccnt 0x4" ::: "memory");
#endif
}
__device__ __forceinline__ void wait_async_0() {
#if MHA_ASYNC
  asm volatile("s_wait_asynccnt 0x0" ::: "memory");
#endif
}

// Build a 16-element f16 A/B fragment from two aligned 8-half LDS chunks.
__device__ __forceinline__ v16h frag16(const _Float16* p0, const _Float16* p1) {
  v8h a = *(const v8h*)p0;
  v8h b = *(const v8h*)p1;
  v16h r;
#pragma unroll
  for (int i = 0; i < 8; ++i) { r[i] = a[i]; r[i + 8] = b[i]; }
  return r;
}

__device__ __forceinline__ v8f wmma_f16(v16h a, v16h b, v8f c) {
  // v_wmma_f32_16x16x32_f16: D = A(16x32) * B(32x16) + C, f32 accumulate
  return __builtin_amdgcn_wmma_f32_16x16x32_f16(false, a, false, b, (short)0, c,
                                                false, false);
}

// ---------------------------------------------------------------------------
// GEMM: out[M,N] = A[M,K] @ W[K,N] + bias  (A converted to f16, f32 accum)
// Block tile 64x128, 128 threads (4 waves 2x2), wave tile 32x64 (2x4 WMMA):
// 6 fragment loads feed 8 WMMAs per k-step.
// SPLIT_HEADS: out stored f16 as [B, H, S, Dh]; else plain row-major [M,N].
// ---------------------------------------------------------------------------
template<typename AT, bool SPLIT_HEADS, typename OT>
__global__ void __launch_bounds__(128)
mha_gemm_wmma(const AT* __restrict__ A, const float* __restrict__ W,
              const float* __restrict__ bias, OT* __restrict__ out,
              int M, int N, int K)
{
  __shared__ alignas(16) _Float16 smA[64 * 32];    // [m][k]
  __shared__ alignas(16) _Float16 smBt[128 * 32];  // [n][k] (N-major for B frags)

  const int tid  = threadIdx.x;
  const int lane = tid & 31;
  const int wave = tid >> 5;
  const int ln   = lane & 15;
  const int hi   = lane >> 4;

  const int mBlk = blockIdx.y * 64;
  const int nBlk = blockIdx.x * 128;
  const int wm   = (wave >> 1) * 32;   // 0 / 32
  const int wn   = (wave & 1) * 64;    // 0 / 64

  v8f acc[2][4] = {};

  for (int k0 = 0; k0 < K; k0 += 32) {
    // prefetch next k-slice into L2/WGP$ while this stage computes
    if (k0 + 32 < K) {
      if (tid < 64)
        __builtin_prefetch(&A[(size_t)(mBlk + tid) * K + (k0 + 32)], 0, 1);
      __builtin_prefetch(&W[(size_t)(k0 + 32 + (tid >> 2)) * N + nBlk +
                            (tid & 3) * 32], 0, 1);
    }
    // stage A tile 64x32 (coalesced rows of 32)
#pragma unroll
    for (int i = 0; i < 16; ++i) {
      int idx = tid + i * 128;
      int r = idx >> 5, c = idx & 31;
      smA[idx] = (_Float16)A[(size_t)(mBlk + r) * K + (k0 + c)];
    }
    // stage W tile 32x128 transposed -> smBt[n][k] (coalesced over n)
#pragma unroll
    for (int i = 0; i < 32; ++i) {
      int idx = tid + i * 128;
      int kk = idx >> 7, n = idx & 127;
      smBt[n * 32 + kk] = (_Float16)W[(size_t)(k0 + kk) * N + (nBlk + n)];
    }
    __syncthreads();

    v16h af[2], bf[4];
#pragma unroll
    for (int mi = 0; mi < 2; ++mi) {
      const _Float16* p = &smA[(wm + 16 * mi + ln) * 32 + 8 * hi];
      af[mi] = frag16(p, p + 16);            // K chunks {8hi..}, {16+8hi..}
    }
#pragma unroll
    for (int ni = 0; ni < 4; ++ni) {
      const _Float16* p = &smBt[(wn + 16 * ni + ln) * 32 + 8 * hi];
      bf[ni] = frag16(p, p + 16);            // K chunks {8hi..}, {16+8hi..}
    }
#pragma unroll
    for (int mi = 0; mi < 2; ++mi)
#pragma unroll
      for (int ni = 0; ni < 4; ++ni)
        acc[mi][ni] = wmma_f16(af[mi], bf[ni], acc[mi][ni]);
    __syncthreads();
  }

  // epilogue: C/D layout -> lane holds col ln, VGPR r -> row r + 8*hi
#pragma unroll
  for (int mi = 0; mi < 2; ++mi)
#pragma unroll
    for (int ni = 0; ni < 4; ++ni)
#pragma unroll
      for (int r = 0; r < 8; ++r) {
        int row = mBlk + wm + 16 * mi + r + 8 * hi;
        int col = nBlk + wn + 16 * ni + ln;
        float v = acc[mi][ni][r] + bias[col];
        if (SPLIT_HEADS) {
          int b = row >> 11;                 // row / SEQ
          int s = row & (SEQ - 1);
          int h = col >> 6;                  // col / D_HEAD
          int d = col & 63;
          out[((((size_t)b * NUM_HEADS + h) * SEQ + s) << 6) + d] = (OT)v;
        } else {
          out[(size_t)row * N + col] = (OT)v;
        }
      }
}

// ---------------------------------------------------------------------------
// Flash attention: block = one (b,h) x 64 query rows; 4 waves x 16 rows.
// K tiles double-buffered via async global->LDS DMA (ASYNCcnt); V staged
// through VGPRs (needs transpose); online softmax; WMMA for QK^T and PV.
// ctx written f16 as [B, S, H*Dh] ready for the output projection GEMM.
// ---------------------------------------------------------------------------
__global__ void __launch_bounds__(128)
mha_flash_wmma(const _Float16* __restrict__ Qm, const _Float16* __restrict__ Km,
               const _Float16* __restrict__ Vm, _Float16* __restrict__ ctx)
{
  __shared__ alignas(16) _Float16 smQ[64 * 64];
  __shared__ alignas(16) _Float16 smK[2][64 * 64];   // [key][d], double-buffered
  __shared__ alignas(16) _Float16 smVt[64 * 64];     // [d][key]
  __shared__ alignas(16) _Float16 smP[4][16 * 64];   // per-wave P tile

  const int tid = threadIdx.x, lane = tid & 31, wave = tid >> 5;
  const int ln = lane & 15, hi = lane >> 4;
  const int bh = blockIdx.x;
  const int b = bh >> 4, h = bh & 15;
  const int q0 = blockIdx.y * 64;

  const _Float16* Qb = Qm + (size_t)bh * SEQ * 64 + (size_t)q0 * 64;
  const _Float16* Kb = Km + (size_t)bh * SEQ * 64;
  const _Float16* Vb = Vm + (size_t)bh * SEQ * 64;

  // Prologue: async-stage Q, then kick off K tile 0 (stays in flight).
  stage_tile(smQ, Qb, tid);
  stage_tile(smK[0], Kb, tid);
  wait_async_le4();        // Q's 4 transfers retired (loads complete in order)
  __syncthreads();

  // Q fragments held in registers for the whole key loop (2 k-steps over Dh)
  v16h qf[2];
#pragma unroll
  for (int ks = 0; ks < 2; ++ks) {
    const _Float16* p = &smQ[(16 * wave + ln) * 64 + 32 * ks + 8 * hi];
    qf[ks] = frag16(p, p + 16);
  }

  float mrow[8], lrow[8];
  v8f oacc[4] = {};
#pragma unroll
  for (int r = 0; r < 8; ++r) { mrow[r] = -1e30f; lrow[r] = 0.f; }

  const int NT = SEQ / 64;   // 32 key tiles
  for (int it = 0; it < NT; ++it) {
    const int kt = it * 64;
    { // stage V tile transposed -> smVt[d][key] (VGPR path: transpose)
      const _Float16* src = Vb + (size_t)kt * 64;
#pragma unroll
      for (int i = 0; i < 32; ++i) {
        int idx = tid + i * 128;
        int key = idx >> 6, d = idx & 63;
        smVt[d * 64 + key] = src[idx];
      }
    }
    // kick off next K tile; the end-of-iteration barrier from it-1 guarantees
    // nobody still reads buffer (it+1)&1.
    if (it + 1 < NT) {
      stage_tile(smK[(it + 1) & 1], Kb + (size_t)(kt + 64) * 64, tid);
      wait_async_le4();      // K(it) done, K(it+1) still in flight
    } else {
      wait_async_0();
    }
    __syncthreads();
    const _Float16* K0 = smK[it & 1];

    // S = scale * Q K^T  (4 key sub-tiles of 16)
    v8f sc[4];
#pragma unroll
    for (int j = 0; j < 4; ++j) {
      v8f s = {};
#pragma unroll
      for (int ks = 0; ks < 2; ++ks) {
        const _Float16* p = &K0[(16 * j + ln) * 64 + 32 * ks + 8 * hi];
        s = wmma_f16(qf[ks], frag16(p, p + 16), s);
      }
#pragma unroll
      for (int r = 0; r < 8; ++r) s[r] *= 0.125f;   // 1/sqrt(64)
      sc[j] = s;
    }

    // online softmax: row reductions across the 16 lanes of each half-wave
    float mnew[8], alpha[8], rs[8];
#pragma unroll
    for (int r = 0; r < 8; ++r) {
      float t = sc[0][r];
#pragma unroll
      for (int j = 1; j < 4; ++j) t = fmaxf(t, sc[j][r]);
      t = fmaxf(t, __shfl_xor(t, 1, 32));
      t = fmaxf(t, __shfl_xor(t, 2, 32));
      t = fmaxf(t, __shfl_xor(t, 4, 32));
      t = fmaxf(t, __shfl_xor(t, 8, 32));
      mnew[r]  = fmaxf(mrow[r], t);
      alpha[r] = __expf(mrow[r] - mnew[r]);
      mrow[r]  = mnew[r];
      rs[r]    = 0.f;
    }
    // P = exp(S - m_new), written to per-wave LDS tile (C-layout -> A-layout)
#pragma unroll
    for (int j = 0; j < 4; ++j)
#pragma unroll
      for (int r = 0; r < 8; ++r) {
        float p = __expf(sc[j][r] - mnew[r]);
        rs[r] += p;
        smP[wave][(r + 8 * hi) * 64 + 16 * j + ln] = (_Float16)p;
      }
#pragma unroll
    for (int r = 0; r < 8; ++r) {
      float t = rs[r];
      t += __shfl_xor(t, 1, 32);
      t += __shfl_xor(t, 2, 32);
      t += __shfl_xor(t, 4, 32);
      t += __shfl_xor(t, 8, 32);
      lrow[r] = lrow[r] * alpha[r] + t;
    }
#pragma unroll
    for (int j = 0; j < 4; ++j)
#pragma unroll
      for (int r = 0; r < 8; ++r)
        oacc[j][r] *= alpha[r];

    // O += P V  (LDS DS ops are in-order within a wave, so smP is visible)
#pragma unroll
    for (int ks = 0; ks < 2; ++ks) {
      const _Float16* pp = &smP[wave][ln * 64 + 32 * ks + 8 * hi];
      v16h pf = frag16(pp, pp + 16);
#pragma unroll
      for (int j = 0; j < 4; ++j) {
        const _Float16* vp = &smVt[(16 * j + ln) * 64 + 32 * ks + 8 * hi];
        oacc[j] = wmma_f16(pf, frag16(vp, vp + 16), oacc[j]);
      }
    }
    __syncthreads();
  }

  // normalize and emit ctx f16 in [B, S, H*Dh]
#pragma unroll
  for (int r = 0; r < 8; ++r) {
    float inv = 1.f / lrow[r];
    int qs = q0 + 16 * wave + r + 8 * hi;
#pragma unroll
    for (int j = 0; j < 4; ++j) {
      int d = 16 * j + ln;
      ctx[((size_t)b * SEQ + qs) * (NUM_HEADS * D_HEAD) + h * 64 + d] =
          (_Float16)(oacc[j][r] * inv);
    }
  }
}

// ---------------------------------------------------------------------------
extern "C" void kernel_launch(void* const* d_in, const int* in_sizes, int n_in,
                              void* d_out, int out_size, void* d_ws, size_t ws_size,
                              hipStream_t stream)
{
  (void)in_sizes; (void)n_in; (void)out_size; (void)ws_size;

  const float* x  = (const float*)d_in[0];
  const float* Wq = (const float*)d_in[1];
  const float* bq = (const float*)d_in[2];
  const float* Wk = (const float*)d_in[3];
  const float* bk = (const float*)d_in[4];
  const float* Wv = (const float*)d_in[5];
  const float* bv = (const float*)d_in[6];
  const float* Wo = (const float*)d_in[7];
  const float* bo = (const float*)d_in[8];
  float* out = (float*)d_out;

  const size_t mat = (size_t)MROWS * D_MODEL;       // 4096*1024 elems
  _Float16* qf = (_Float16*)d_ws;                   // [B,H,S,Dh] f16
  _Float16* kf = qf + mat;
  _Float16* vf = kf + mat;
  _Float16* cf = vf + mat;                          // ctx [B,S,H*Dh] f16

  dim3 blk(128);
  dim3 gproj(D_MODEL / 128, MROWS / 64);            // (8, 64)

  mha_gemm_wmma<float, true, _Float16>
      <<<gproj, blk, 0, stream>>>(x, Wq, bq, qf, MROWS, D_MODEL, D_MODEL);
  mha_gemm_wmma<float, true, _Float16>
      <<<gproj, blk, 0, stream>>>(x, Wk, bk, kf, MROWS, D_MODEL, D_MODEL);
  mha_gemm_wmma<float, true, _Float16>
      <<<gproj, blk, 0, stream>>>(x, Wv, bv, vf, MROWS, D_MODEL, D_MODEL);

  dim3 gfa(BATCH * NUM_HEADS, SEQ / 64);            // (32, 32)
  mha_flash_wmma<<<gfa, blk, 0, stream>>>(qf, kf, vf, cf);

  mha_gemm_wmma<_Float16, false, float>
      <<<gproj, blk, 0, stream>>>(cf, Wo, bo, out, MROWS, D_MODEL, D_MODEL);
}
